// KnowledgeGraph_67997922230377
// MI455X (gfx1250) — matrix operand order repaired and checked by reading, not compiled, not run on hardware
//
#include <hip/hip_runtime.h>
#include <hip/hip_bf16.h>
#include <stdint.h>

typedef __attribute__((ext_vector_type(16))) __bf16 v16bf;
typedef __attribute__((ext_vector_type(8)))  float  v8f;

#define N_ENT   100000
#define DIM     256
#define NT      4
#define NH      32
#define THBITS  128      // T*H bits per code
#define BQ      64
#define TOPK    10
#define NCAND   10000

// ---- workspace layout (bytes) ----
#define OFF_TARGET  0u            // 64*256*4       = 65536
#define OFF_THASH   65536u        // 64*4*4         = 1024
#define OFF_TAUM    66560u        // 64*2*4         = 512
#define OFF_BMAT    67584u        // 12*8*32*16*2   = 98304  (bf16, WMMA-B fragment order)
#define OFF_GHIST   165888u       // 64*129*4       = 33024
#define OFF_DIST    198912u       // 64*100000 u8   = 6400000
#define OFF_SCORES  6598912u      // 64*100000 f32  = 25600000  (end ~32.2 MB)

#define BMAT_BYTES  98304

__device__ __forceinline__ unsigned short f32_to_bf16_rne(float f) {
    unsigned u = __float_as_uint(f);
    u += 0x7FFFu + ((u >> 16) & 1u);
    return (unsigned short)(u >> 16);
}

// ---------------- P1: target = h*r, query hash codes (ballot-packed) --------
__global__ __launch_bounds__(256) void kg_p1_target_thash(
    const int* __restrict__ head, const int* __restrict__ rel,
    const float* __restrict__ ent, const float* __restrict__ remb,
    const float* __restrict__ hproj,
    float* __restrict__ target, unsigned* __restrict__ thash)
{
    __shared__ float tg[DIM];
    int b = blockIdx.x, t = threadIdx.x;
    int h = head[b], r = rel[b];
    float v = ent[(size_t)h * DIM + t] * remb[(size_t)r * DIM + t];
    tg[t] = v;
    target[b * DIM + t] = v;
    __syncthreads();
    if (t < THBITS) {
        int tt = t >> 5, hh = t & 31;
        const float* hp = hproj + (size_t)tt * DIM * NH + hh;
        float acc = 0.f;
        for (int d = 0; d < DIM; ++d) acc += tg[d] * hp[d * NH];
        unsigned long long m = __ballot(acc > 0.0f);   // wave32: low 32 bits
        if ((t & 31) == 0) thash[b * 4 + (t >> 5)] = (unsigned)m;
    }
}

// ---------------- P2: build bf16 B matrix in per-lane WMMA-B fragment order -
// Logical B[k][n]: n<128 -> hash_proj[t=n>>5][k][h=n&31]; n>=128 -> target[n-128][k]
// Fragment layout: flat = ((j*8+s)*32+lane)*16+i ; k = s*32 + (lane>=16?8:0) + (i&7) + (i>>3)*16 ;
// n = j*16 + (lane&15)   (B rows: lanes 0-15 hold K {0..7,16..23}, lanes 16-31 K {8..15,24..31})
__global__ __launch_bounds__(256) void kg_p2_bmat(
    const float* __restrict__ hproj, const float* __restrict__ target,
    unsigned short* __restrict__ bmat)
{
    int f = blockIdx.x * 256 + threadIdx.x;   // 0..49151
    int i    = f & 15;
    int lane = (f >> 4) & 31;
    int s    = (f >> 9) & 7;
    int j    = f >> 12;                       // 0..11
    int hi = lane >> 4, lm = lane & 15;
    int n  = j * 16 + lm;
    int kk = s * 32 + hi * 8 + (i & 7) + (i >> 3) * 16;
    float v;
    if (n < THBITS) {
        int tt = n >> 5, hh = n & 31;
        v = hproj[((size_t)tt * DIM + kk) * NH + hh];
    } else {
        v = target[(n - THBITS) * DIM + kk];
    }
    bmat[f] = f32_to_bf16_rne(v);
}

// ---------------- P3: fused WMMA GEMM + hash-pack + Hamming + score store ---
// Dynamic LDS buffer (98304 B): holds the shared B fragments during the GEMM,
// then is reused as the per-wave C transpose stage (8*16*100*4 = 51200 B).
__global__ __launch_bounds__(256) void kg_p3_gemm(
    const float* __restrict__ ent, const unsigned short* __restrict__ bmat,
    const unsigned* __restrict__ thash,
    float* __restrict__ scores, unsigned char* __restrict__ dist)
{
    extern __shared__ __align__(16) unsigned char smem[];   // BMAT_BYTES
    __shared__ unsigned ths[BQ * 4];
    int tid = threadIdx.x;
    ths[tid] = thash[tid];               // 256 == 64*4

    // ---- async copy B: global -> LDS, 24 x b128 per thread (98304 B total) ----
    {
        unsigned smemBase = (unsigned)(size_t)(void*)smem;  // LDS offset = addr[31:0]
        const char* gb = (const char*)bmat;
        #pragma unroll
        for (int it = 0; it < 24; ++it) {
            unsigned off = (unsigned)((it * 256 + tid) * 16);
            unsigned ldsOff = smemBase + off;
            unsigned long long ga = (unsigned long long)(size_t)(gb + off);
            asm volatile("global_load_async_to_lds_b128 %0, %1, off"
                         :: "v"(ldsOff), "v"(ga) : "memory");
        }
        asm volatile("s_wait_asynccnt 0x0" ::: "memory");
    }
    __syncthreads();    // B visible to all waves (also covers ths)

    int w = tid >> 5, lane = tid & 31;
    int hi = lane >> 4, lm = lane & 15;
    int n0 = blockIdx.x * 128 + w * 16;
    int arow = n0 + lm; if (arow > N_ENT - 1) arow = N_ENT - 1;   // clamp tail rows
    const float* ap = ent + (size_t)arow * DIM + hi * 8;
    const unsigned short* bsh = (const unsigned short*)smem;

    v8f acc[12];
    v8f zero = {0.f,0.f,0.f,0.f,0.f,0.f,0.f,0.f};
    #pragma unroll
    for (int j = 0; j < 12; ++j) acc[j] = zero;

    #pragma unroll
    for (int s = 0; s < 8; ++s) {
        // A fragment: lane holds K = {0..7,16..23} (+8 if upper half-wave)
        const float4* a4 = (const float4*)(ap + s * 32);
        float4 a0 = a4[0], a1 = a4[1], a2 = a4[4], a3 = a4[5];
        float av[16] = {a0.x,a0.y,a0.z,a0.w, a1.x,a1.y,a1.z,a1.w,
                        a2.x,a2.y,a2.z,a2.w, a3.x,a3.y,a3.z,a3.w};
        union { unsigned short u[16]; v16bf v; } A;
        #pragma unroll
        for (int i = 0; i < 16; ++i) A.u[i] = f32_to_bf16_rne(av[i]);
        #pragma unroll
        for (int j = 0; j < 12; ++j) {
            const v16bf* bp = reinterpret_cast<const v16bf*>(
                bsh + (((j * 8 + s) * 32 + lane) << 4));
            v16bf B = *bp;
            acc[j] = __builtin_amdgcn_wmma_f32_16x16x32_bf16(
                false, A.v, false, B, (short)0, acc[j], false, false);
        }
    }

    __syncthreads();    // all waves done reading B; reuse LDS as C stage
    float* cstf = (float*)smem;             // [8][16][100] padded rows
#define CST(row, c) cstf[((w * 16 + (row)) * 100) + (c)]

    // ---- phase 1: stage cols 0..95, build hash words 0..2 ----
    #pragma unroll
    for (int j = 0; j < 6; ++j)
        #pragma unroll
        for (int r = 0; r < 8; ++r)
            CST(r + hi * 8, j * 16 + lm) = acc[j][r];

    unsigned wa = 0, wb = 0;
    if (hi == 0) {
        for (int c = 0; c < 32; ++c) wa |= (CST(lm, c)      > 0.f) ? (1u << c) : 0u;
        for (int c = 0; c < 32; ++c) wb |= (CST(lm, 32 + c) > 0.f) ? (1u << c) : 0u;
    } else {
        for (int c = 0; c < 32; ++c) wa |= (CST(lm, 64 + c) > 0.f) ? (1u << c) : 0u;
    }

    // ---- phase 2: stage cols 96..191, finish word 3, emit scores ----
    #pragma unroll
    for (int j = 6; j < 12; ++j)
        #pragma unroll
        for (int r = 0; r < 8; ++r)
            CST(r + hi * 8, (j - 6) * 16 + lm) = acc[j][r];

    int n = n0 + lm;
    if (hi == 1) {
        for (int c = 0; c < 32; ++c) wb |= (CST(lm, c) > 0.f) ? (1u << c) : 0u;
    }
    {   // scores: local cols 32..95 <-> b = 0..63 ; coalesced per 16-lane group
        int bb0 = hi * 32;
        for (int q = 0; q < 32; ++q) {
            float sv = CST(lm, 32 + bb0 + q);
            if (n < N_ENT) scores[(size_t)(bb0 + q) * N_ENT + n] = sv;
        }
    }
#undef CST
    // exchange hash words across half-waves -> every lane has all 4 words of its row
    unsigned xa = (unsigned)__shfl_xor((int)wa, 16, 32);
    unsigned xb = (unsigned)__shfl_xor((int)wb, 16, 32);
    unsigned h0 = hi ? xa : wa;
    unsigned h1 = hi ? xb : wb;
    unsigned h2 = hi ? wa : xa;
    unsigned h3 = hi ? wb : xb;

    if (n < N_ENT) {
        int bb0 = hi * 32;
        for (int q = 0; q < 32; ++q) {
            int b = bb0 + q;
            unsigned d = __popc(h0 ^ ths[b * 4 + 0]) + __popc(h1 ^ ths[b * 4 + 1])
                       + __popc(h2 ^ ths[b * 4 + 2]) + __popc(h3 ^ ths[b * 4 + 3]);
            dist[(size_t)b * N_ENT + n] = (unsigned char)d;
        }
    }
}

// ---------------- P4: per-query Hamming histograms (129 bins) ---------------
__global__ __launch_bounds__(256) void kg_p4_hist(
    const unsigned char* __restrict__ dist, unsigned* __restrict__ ghist)
{
    __shared__ unsigned h[129];
    int tid = threadIdx.x;
    int b = blockIdx.y;
    if (tid < 129) h[tid] = 0;
    __syncthreads();
    int base = blockIdx.x * 4096;
    const unsigned char* db = dist + (size_t)b * N_ENT;
    for (int i = 0; i < 16; ++i) {
        int nn = base + i * 256 + tid;
        if (nn < N_ENT) atomicAdd(&h[db[nn]], 1u);
    }
    __syncthreads();
    if (tid < 129) atomicAdd(&ghist[b * 129 + tid], h[tid]);
}

// ---------------- P5: threshold tau + tie budget m per query ----------------
__global__ void kg_p5_thresh(const unsigned* __restrict__ ghist,
                             unsigned* __restrict__ taum)
{
    int b = threadIdx.x;
    if (b >= BQ) return;
    unsigned cum = 0, tau = 128, m = 0;
    for (int d = 0; d <= 128; ++d) {
        unsigned c = ghist[b * 129 + d];
        if (cum + c >= (unsigned)NCAND) { tau = (unsigned)d; m = NCAND - cum; break; }
        cum += c;
    }
    taum[b * 2 + 0] = tau;
    taum[b * 2 + 1] = m;
}

// ---------------- P6: per-query candidate scan + top-10 ---------------------
__global__ __launch_bounds__(256) void kg_p6_select(
    const unsigned char* __restrict__ dist, const float* __restrict__ scores,
    const unsigned* __restrict__ taum, float* __restrict__ out)
{
    __shared__ unsigned waveTot[8];
    __shared__ float ms[2560];
    __shared__ int   mi[2560];
    __shared__ float bs[256];
    __shared__ int   bi[256];
    __shared__ int   bslot[256];

    int tid = threadIdx.x, b = blockIdx.x;
    int w = tid >> 5, lane = tid & 31;
    unsigned tau = taum[b * 2 + 0], m = taum[b * 2 + 1];
    const unsigned char* db = dist + (size_t)b * N_ENT;
    const float* sb = scores + (size_t)b * N_ENT;

    float ts[10]; int ti[10];
    #pragma unroll
    for (int q = 0; q < 10; ++q) { ts[q] = -__builtin_inff(); ti[q] = 0x7fffffff; }

    unsigned carry = 0;                         // index-ordered tie count so far
    const int CH = (N_ENT + 255) / 256;         // 391
    for (int c = 0; c < CH; ++c) {
        int n = c * 256 + tid;
        bool valid = n < N_ENT;
        unsigned d = valid ? (unsigned)db[n] : 255u;
        bool tie = (d == tau);
        unsigned mask = (unsigned)__ballot(tie);
        unsigned rk = __popc(mask & ((1u << lane) - 1u));
        unsigned wt = __popc(mask);
        __syncthreads();                        // protect waveTot reuse
        if (lane == 0) waveTot[w] = wt;
        __syncthreads();
        unsigned pre = 0, tot = 0;
        #pragma unroll
        for (int q = 0; q < 8; ++q) { unsigned v = waveTot[q]; if (q < w) pre += v; tot += v; }
        bool inc = valid && ((d < tau) || (tie && (carry + pre + rk) < m));
        carry += tot;
        if (inc) {
            float s = sb[n];
            if (s > ts[9] || (s == ts[9] && n < ti[9])) {
                ts[9] = s; ti[9] = n;
                #pragma unroll
                for (int q = 9; q > 0; --q) {
                    bool sw = (ts[q] > ts[q-1]) || (ts[q] == ts[q-1] && ti[q] < ti[q-1]);
                    if (sw) { float tf = ts[q]; ts[q] = ts[q-1]; ts[q-1] = tf;
                              int tn = ti[q];  ti[q] = ti[q-1]; ti[q-1] = tn; }
                }
            }
        }
    }
    #pragma unroll
    for (int q = 0; q < 10; ++q) { ms[tid * 10 + q] = ts[q]; mi[tid * 10 + q] = ti[q]; }
    __syncthreads();

    for (int r = 0; r < TOPK; ++r) {
        float bestv = -__builtin_inff(); int besti = 0x7fffffff; int bsl = 0;
        #pragma unroll
        for (int q = 0; q < 10; ++q) {
            float v = ms[tid * 10 + q]; int ix = mi[tid * 10 + q];
            if (v > bestv || (v == bestv && ix < besti)) { bestv = v; besti = ix; bsl = tid * 10 + q; }
        }
        bs[tid] = bestv; bi[tid] = besti; bslot[tid] = bsl;
        __syncthreads();
        if (tid == 0) {
            float gv = -__builtin_inff(); int gi = 0x7fffffff; int gs = 0;
            for (int t2 = 0; t2 < 256; ++t2) {
                if (bs[t2] > gv || (bs[t2] == gv && bi[t2] < gi)) {
                    gv = bs[t2]; gi = bi[t2]; gs = bslot[t2];
                }
            }
            out[b * TOPK + r] = (float)gi;              // indices (flat, first)
            out[BQ * TOPK + b * TOPK + r] = gv;         // scores (second)
            ms[gs] = -__builtin_inff(); mi[gs] = 0x7fffffff;   // remove winner
        }
        __syncthreads();
    }
}

extern "C" void kernel_launch(void* const* d_in, const int* in_sizes, int n_in,
                              void* d_out, int out_size, void* d_ws, size_t ws_size,
                              hipStream_t stream)
{
    (void)in_sizes; (void)n_in; (void)out_size; (void)ws_size;
    const int*   head  = (const int*)d_in[0];
    const int*   rel   = (const int*)d_in[1];
    const float* ent   = (const float*)d_in[2];
    const float* remb  = (const float*)d_in[3];
    const float* hproj = (const float*)d_in[4];
    // d_in[5] = k (always 10; CAND/K baked in at compile time)

    char* ws = (char*)d_ws;
    float*          target = (float*)(ws + OFF_TARGET);
    unsigned*       thash  = (unsigned*)(ws + OFF_THASH);
    unsigned*       taum   = (unsigned*)(ws + OFF_TAUM);
    unsigned short* bmat   = (unsigned short*)(ws + OFF_BMAT);
    unsigned*       ghist  = (unsigned*)(ws + OFF_GHIST);
    unsigned char*  dist   = (unsigned char*)(ws + OFF_DIST);
    float*          scores = (float*)(ws + OFF_SCORES);

    kg_p1_target_thash<<<BQ, 256, 0, stream>>>(head, rel, ent, remb, hproj, target, thash);
    kg_p2_bmat<<<192, 256, 0, stream>>>(hproj, target, bmat);
    hipMemsetAsync(ghist, 0, BQ * 129 * sizeof(unsigned), stream);
    kg_p3_gemm<<<(N_ENT + 127) / 128, 256, BMAT_BYTES, stream>>>(ent, bmat, thash, scores, dist);
    kg_p4_hist<<<dim3((N_ENT + 4095) / 4096, BQ), 256, 0, stream>>>(dist, ghist);
    kg_p5_thresh<<<1, 64, 0, stream>>>(ghist, taum);
    kg_p6_select<<<BQ, 256, 0, stream>>>(dist, scores, taum, (float*)d_out);
}
